// GLOWFlow_65137474011165
// MI455X (gfx1250) — matrix operand order repaired and checked by reading, not compiled, not run on hardware
//
#include <hip/hip_runtime.h>
#include <hip/hip_bf16.h>

typedef __attribute__((ext_vector_type(16))) _Float16 v16h;
typedef __attribute__((ext_vector_type(8)))  float    v8f;

#define NF  8
#define HID 64
#define BLK 256

__device__ __forceinline__ float relu_f(float x) { return fmaxf(x, 0.0f); }

__global__ __launch_bounds__(BLK)
void glow_wmma_kernel(
    const float* __restrict__ g_eps,    // [N,3]
    const float* __restrict__ g_Ltril,  // [3,3]
    const float* __restrict__ g_bmean,  // [3]
    const float* __restrict__ g_anls,   // [NF,3]
    const float* __restrict__ g_ansh,   // [NF,3]
    const float* __restrict__ g_perm,   // [NF,3,3]
    const float* __restrict__ g_sign,   // [NF,3]
    const float* __restrict__ g_lul,    // [NF,3,3]
    const float* __restrict__ g_luu,    // [NF,3,3]
    const float* __restrict__ g_luls,   // [NF,3]
    const float* __restrict__ g_W1,     // [NF,64,1]
    const float* __restrict__ g_b1,     // [NF,64]
    const float* __restrict__ g_W2,     // [NF,64,64]
    const float* __restrict__ g_b2,     // [NF,64]
    const float* __restrict__ g_W3,     // [NF,4,64]
    const float* __restrict__ g_b3,     // [NF,4]
    float* __restrict__ g_out,          // [N*3] z_t, then [N] log_det
    int N)
{
  // Shared across block (barrier-fenced per flow):
  __shared__ _Float16 sW2[HID * HID];          // W2 as f16, [n][k] row-major
  __shared__ float    sW1[HID], sB1[HID], sB2[HID];
  // Wave-private regions (relies on in-order LDS within a wave, no barriers):
  __shared__ float    sZ1[8][32];              // z1 per sample
  __shared__ _Float16 sH2[8][16 * HID];        // one 16x64 H2 tile per wave
  __shared__ float    sOut[8][2][16][4];       // coupling-net outputs

  const int tid  = threadIdx.x;
  const int wv   = tid >> 5;    // wave in block
  const int lane = tid & 31;
  const int hf   = lane >> 4;   // half-wave (0/1)
  const int idx  = lane & 15;
  int s = blockIdx.x * BLK + tid;
  if (s >= N) s = N - 1;        // clamp keeps EXEC full for WMMA; dup writes are identical

  // ---- base distribution: Lt = tril(L)+1e-6 I; cov = Lt Lt^T; L = chol(cov) ----
  float a00 = g_Ltril[0] + 1e-6f;
  float a10 = g_Ltril[3];
  float a11 = g_Ltril[4] + 1e-6f;
  float a20 = g_Ltril[6];
  float a21 = g_Ltril[7];
  float a22 = g_Ltril[8] + 1e-6f;
  float c00 = a00 * a00;
  float c10 = a10 * a00;
  float c11 = a10 * a10 + a11 * a11;
  float c20 = a20 * a00;
  float c21 = a20 * a10 + a21 * a11;
  float c22 = a20 * a20 + a21 * a21 + a22 * a22;
  float l00 = sqrtf(c00);
  float l10 = c10 / l00;
  float l20 = c20 / l00;
  float l11 = sqrtf(c11 - l10 * l10);
  float l21 = (c21 - l20 * l10) / l11;
  float l22 = sqrtf(c22 - l20 * l20 - l21 * l21);

  float e0 = g_eps[3 * s + 0], e1 = g_eps[3 * s + 1], e2 = g_eps[3 * s + 2];
  float z0 = g_bmean[0] + e0 * l00;
  float z1 = g_bmean[1] + e0 * l10 + e1 * l11;
  float z2 = g_bmean[2] + e0 * l20 + e1 * l21 + e2 * l22;
  float ld = 0.0f;

  for (int f = 0; f < NF; ++f) {
    __syncthreads();  // protect sW2/sW1/sB1/sB2 from previous flow's readers
    for (int i = tid; i < HID * HID; i += BLK)
      sW2[i] = (_Float16)g_W2[f * HID * HID + i];
    if (tid < HID) {
      sW1[tid] = g_W1[f * HID + tid];
      sB1[tid] = g_b1[f * HID + tid];
      sB2[tid] = g_b2[f * HID + tid];
    }
    __syncthreads();

    // ---- actnorm ----
    {
      float ls0 = g_anls[3 * f + 0], ls1 = g_anls[3 * f + 1], ls2 = g_anls[3 * f + 2];
      float sh0 = g_ansh[3 * f + 0], sh1 = g_ansh[3 * f + 1], sh2 = g_ansh[3 * f + 2];
      z0 = expf(ls0) * (z0 + sh0);
      z1 = expf(ls1) * (z1 + sh1);
      z2 = expf(ls2) * (z2 + sh2);
      ld += ls0 + ls1 + ls2;
    }

    // ---- invertible 1x1 conv (PLU): W = P (tril(l,-1)+I) (triu(u,1)+diag(s*exp(lls))) ----
    {
      const float* lm = g_lul + 9 * f;
      const float* um = g_luu + 9 * f;
      const float* Pm = g_perm + 9 * f;
      float lls0 = g_luls[3 * f + 0], lls1 = g_luls[3 * f + 1], lls2 = g_luls[3 * f + 2];
      float d0 = g_sign[3 * f + 0] * expf(lls0);
      float d1 = g_sign[3 * f + 1] * expf(lls1);
      float d2 = g_sign[3 * f + 2] * expf(lls2);
      float u01 = um[1], u02 = um[2], u12 = um[5];
      float l10m = lm[3], l20m = lm[6], l21m = lm[7];
      // M = Lw * U
      float m00 = d0,        m01 = u01,                  m02 = u02;
      float m10 = l10m * d0, m11 = l10m * u01 + d1,      m12 = l10m * u02 + u12;
      float m20 = l20m * d0, m21 = l20m * u01 + l21m * d1,
            m22 = l20m * u02 + l21m * u12 + d2;
      // W = P * M (P is a permutation, do the dense 3x3 anyway)
      float W00 = Pm[0] * m00 + Pm[1] * m10 + Pm[2] * m20;
      float W01 = Pm[0] * m01 + Pm[1] * m11 + Pm[2] * m21;
      float W02 = Pm[0] * m02 + Pm[1] * m12 + Pm[2] * m22;
      float W10 = Pm[3] * m00 + Pm[4] * m10 + Pm[5] * m20;
      float W11 = Pm[3] * m01 + Pm[4] * m11 + Pm[5] * m21;
      float W12 = Pm[3] * m02 + Pm[4] * m12 + Pm[5] * m22;
      float W20 = Pm[6] * m00 + Pm[7] * m10 + Pm[8] * m20;
      float W21 = Pm[6] * m01 + Pm[7] * m11 + Pm[8] * m21;
      float W22 = Pm[6] * m02 + Pm[7] * m12 + Pm[8] * m22;
      float t0 = W00 * z0 + W01 * z1 + W02 * z2;
      float t1 = W10 * z0 + W11 * z1 + W12 * z2;
      float t2 = W20 * z0 + W21 * z1 + W22 * z2;
      z0 = t0; z1 = t1; z2 = t2;
      ld += lls0 + lls1 + lls2;
    }

    // ---- affine coupling: MLP(z1) -> shift/log_scale for (z2a,z2b) ----
    sZ1[wv][lane] = z0;

    // W3 B-fragments (B layout: k = c*32 + ((v&4)<<2) + 8*hf + 2*(v&3) + e; n = idx, zero-padded n>=4)
    v16h bw3[2];
    #pragma unroll
    for (int c = 0; c < 2; ++c)
      #pragma unroll
      for (int v = 0; v < 8; ++v)
        #pragma unroll
        for (int e = 0; e < 2; ++e) {
          int k = c * 32 + ((v & 4) << 2) + hf * 8 + (v & 3) * 2 + e;
          float w = (idx < 4) ? g_W3[f * 4 * HID + idx * HID + k] : 0.0f;
          bw3[c][2 * v + e] = (_Float16)w;
        }

    #pragma unroll
    for (int t = 0; t < 2; ++t) {
      // H1 A-fragments built in place: h1[m][k] = relu(z1[m]*w1[k]+b1[k])
      // (A layout: k = c*32 + ((v&4)<<2) + 8*hf + 2*(v&3) + e; m = idx)
      v16h ah[2];
      float z1t = sZ1[wv][t * 16 + idx];
      #pragma unroll
      for (int c = 0; c < 2; ++c)
        #pragma unroll
        for (int v = 0; v < 8; ++v)
          #pragma unroll
          for (int e = 0; e < 2; ++e) {
            int k = c * 32 + ((v & 4) << 2) + hf * 8 + (v & 3) * 2 + e;
            ah[c][2 * v + e] = (_Float16)relu_f(z1t * sW1[k] + sB1[k]);
          }

      // hidden layer: H2 = relu(H1 @ W2^T + b2), N=64 in 4 tiles, K=64 in 2 chunks
      #pragma unroll
      for (int nt = 0; nt < 4; ++nt) {
        const int n = nt * 16 + idx;
        v16h bb[2];
        #pragma unroll
        for (int c = 0; c < 2; ++c)
          #pragma unroll
          for (int v = 0; v < 8; ++v)
            #pragma unroll
            for (int e = 0; e < 2; ++e) {
              int k = c * 32 + ((v & 4) << 2) + hf * 8 + (v & 3) * 2 + e;
              bb[c][2 * v + e] = sW2[n * HID + k];   // B[k][n] = W2[n][k]
            }
        v8f acc = {};
        acc = __builtin_amdgcn_wmma_f32_16x16x32_f16(false, ah[0], false, bb[0],
                                                     (short)0, acc, false, false);
        acc = __builtin_amdgcn_wmma_f32_16x16x32_f16(false, ah[1], false, bb[1],
                                                     (short)0, acc, false, false);
        float bias = sB2[n];
        #pragma unroll
        for (int r = 0; r < 8; ++r) {
          int m = r + 8 * hf;                        // D layout: row = r + 8*hf
          sH2[wv][m * HID + n] = (_Float16)relu_f(acc[r] + bias);
        }
      }

      // output layer: OUT = H2 @ W3^T + b3 (columns 0..3 valid)
      v16h ao[2];
      #pragma unroll
      for (int c = 0; c < 2; ++c)
        #pragma unroll
        for (int v = 0; v < 8; ++v)
          #pragma unroll
          for (int e = 0; e < 2; ++e) {
            int k = c * 32 + ((v & 4) << 2) + hf * 8 + (v & 3) * 2 + e;
            ao[c][2 * v + e] = sH2[wv][idx * HID + k];
          }
      v8f acc = {};
      acc = __builtin_amdgcn_wmma_f32_16x16x32_f16(false, ao[0], false, bw3[0],
                                                   (short)0, acc, false, false);
      acc = __builtin_amdgcn_wmma_f32_16x16x32_f16(false, ao[1], false, bw3[1],
                                                   (short)0, acc, false, false);
      if (idx < 4) {
        float b3v = g_b3[4 * f + idx];
        #pragma unroll
        for (int r = 0; r < 8; ++r) {
          int m = r + 8 * hf;
          sOut[wv][t][m][idx] = acc[r] + b3v;
        }
      }
    }

    // apply coupling to this lane's own sample (tile = hf, row = idx)
    {
      float sh0 = sOut[wv][hf][idx][0];
      float sh1 = sOut[wv][hf][idx][1];
      float lg0 = sOut[wv][hf][idx][2];
      float lg1 = sOut[wv][hf][idx][3];
      z1 = z1 * expf(lg0) + sh0;
      z2 = z2 * expf(lg1) + sh1;
      ld += lg0 + lg1;
    }
  }

  // final transform: z_t = [z0, exp(z1), exp(z2)], log_det += z1 + z2
  g_out[3 * s + 0] = z0;
  g_out[3 * s + 1] = expf(z1);
  g_out[3 * s + 2] = expf(z2);
  g_out[3 * N + s] = ld + z1 + z2;
}

extern "C" void kernel_launch(void* const* d_in, const int* in_sizes, int n_in,
                              void* d_out, int out_size, void* d_ws, size_t ws_size,
                              hipStream_t stream) {
  const float* eps    = (const float*)d_in[0];
  const float* Ltril  = (const float*)d_in[1];
  const float* bmean  = (const float*)d_in[2];
  const float* anls   = (const float*)d_in[3];
  const float* ansh   = (const float*)d_in[4];
  const float* perm   = (const float*)d_in[5];
  const float* sgn    = (const float*)d_in[6];
  const float* lul    = (const float*)d_in[7];
  const float* luu    = (const float*)d_in[8];
  const float* luls   = (const float*)d_in[9];
  const float* W1     = (const float*)d_in[10];
  const float* b1     = (const float*)d_in[11];
  const float* W2     = (const float*)d_in[12];
  const float* b2     = (const float*)d_in[13];
  const float* W3     = (const float*)d_in[14];
  const float* b3     = (const float*)d_in[15];
  float* out = (float*)d_out;

  int N = in_sizes[0] / 3;
  int blocks = (N + BLK - 1) / BLK;
  glow_wmma_kernel<<<blocks, BLK, 0, stream>>>(
      eps, Ltril, bmean, anls, ansh, perm, sgn, lul, luu, luls,
      W1, b1, W2, b2, W3, b3, out, N);
}
